// ContextualMemoryBank_84310208020755
// MI455X (gfx1250) — matrix operand order repaired and checked by reading, not compiled, not run on hardware
//
#include <hip/hip_runtime.h>

// ============================================================================
// ContextualMemoryBank for MI455X (gfx1250, wave32, WMMA f16->f32, TDM async)
//
//   1. gnn_layer x3   : h = LN(h + relu((adj@h)@W + b))      [WMMA, PK2 LDS]
//   2. qkv            : q/k/v = h@W + b (f16; v transposed)  [WMMA]
//   3. attn           : flash attention per (b,head,16 rows) [WMMA + online softmax]
//   4. proj           : h = ctx@Wo + bo                      [WMMA]
//   5. query          : query = mean_n(h)@Wq + b (f16)
//   6. logits         : logits[256,S] f16 + split stats      [WMMA + TDM dbl-buf]
//   7. stats combine  : global softmax max / 1/denominator
//   8. retrieve       : part = softmax(logits)@values        [WMMA + TDM dbl-buf]
//   9. combine        : split-K fixup -> d_out f32 [256,512]
// ============================================================================

typedef _Float16 f16;
typedef __attribute__((ext_vector_type(2)))  _Float16     h2;
typedef __attribute__((ext_vector_type(16))) _Float16     v16h;
typedef __attribute__((ext_vector_type(8)))  float        v8f;
typedef __attribute__((ext_vector_type(8)))  unsigned int v8u;
typedef __attribute__((ext_vector_type(2)))  float        f32x2;
typedef __attribute__((ext_vector_type(4)))  unsigned int u32x4;
typedef __attribute__((ext_vector_type(8)))  int          i32x8;
typedef __attribute__((ext_vector_type(4)))  int          i32x4;

#define DEV __device__ __forceinline__

static constexpr int N_ = 512, D_ = 256;
static constexpr int S_ = 262144, KD_ = 256, MD_ = 512;

DEV v8f wmma16(v16h a, v16h b, v8f c) {
  return __builtin_amdgcn_wmma_f32_16x16x32_f16(false, a, false, b, (short)0, c,
                                                false, false);
}

// pack two f32 -> one dword holding 2 f16 (v_cvt_pk_f16_f32 path)
DEV unsigned int pk2(float x, float y) {
  return __builtin_bit_cast(unsigned int, __builtin_amdgcn_cvt_pkrtz(x, y));
}

// ---------------------------------------------------------------------------
// Fragment loaders: every VGPR filled with a single 32-bit load.
// ---------------------------------------------------------------------------
// A fragment: 16x32 f16 tile at src (row-major, leading dim `ld`, even).
DEV v16h load_a(const f16* src, int ld, int lane) {
  const unsigned int* p = (const unsigned int*)(src + (lane & 15) * ld);
  int kh = (lane >> 4) * 8;  // pair units
  v8u u;
#pragma unroll
  for (int v = 0; v < 8; ++v) u[v] = p[kh + v];
  return __builtin_bit_cast(v16h, u);
}

// B fragment of transposed f16 source: B[k][n] = src[n*ld + k] (src NxK row-major)
DEV v16h load_bT(const f16* src, int ld, int lane) {
  const unsigned int* p = (const unsigned int*)(src + (lane & 15) * ld);
  int ko = (lane >> 4) * 8;  // pair units
  v8u u;
#pragma unroll
  for (int v = 0; v < 8; ++v) u[v] = p[ko + v];
  return __builtin_bit_cast(v16h, u);
}

// B fragment of transposed f32 source (TDM-loaded LDS slab): pairs -> cvt_pk
DEV v16h load_bT_f32(const float* src, int ld, int lane) {
  const f32x2* p = (const f32x2*)(src + (lane & 15) * ld);
  int ko = (lane >> 4) * 8;
  v8u u;
#pragma unroll
  for (int v = 0; v < 8; ++v) {
    f32x2 xy = p[ko + v];
    u[v] = pk2(xy.x, xy.y);
  }
  return __builtin_bit_cast(v16h, u);
}

// B fragment from row-major f32 KxN slab (TDM-loaded): strided pair + cvt_pk
DEV v16h load_b_f32(const float* src, int ld, int lane) {
  int col = lane & 15;
  int ko  = (lane >> 4) * 16;
  v8u u;
#pragma unroll
  for (int v = 0; v < 8; ++v)
    u[v] = pk2(src[(ko + 2 * v) * ld + col], src[(ko + 2 * v + 1) * ld + col]);
  return __builtin_bit_cast(v16h, u);
}

// B fragment from PK2 LDS layout: element (k,c) in word [(k>>1)*ldc + c], half k&1
DEV v16h load_b_pk2(const unsigned int* src, int ldc, int lane) {
  int col = lane & 15;
  int ko  = (lane >> 4) * 8;
  v8u u;
#pragma unroll
  for (int v = 0; v < 8; ++v) u[v] = src[(ko + v) * ldc + col];
  return __builtin_bit_cast(v16h, u);
}

// Stage row-major f32 [rows x COLS] (row stride gstride) -> PK2 dwords in LDS.
template <int COLS>
DEV void stage_b_pk2(unsigned int* sB, const float* g, int gstride, int rows,
                     int tid, int nthr) {
  for (int j = tid; j < (rows / 2) * COLS; j += nthr) {
    int pr = j / COLS, c = j % COLS;
    sB[j] = pk2(g[(size_t)(2 * pr) * gstride + c],
                g[(size_t)(2 * pr + 1) * gstride + c]);
  }
}

// Stage row-major f32 [rows x W] -> row-major f16 (as dwords) in LDS.
template <int W>
DEV void stage_a(unsigned int* sA, const float* g, int gstride, int rows,
                 int tid, int nthr) {
  for (int j = tid; j < rows * (W / 2); j += nthr) {
    int r = j / (W / 2), cp = j % (W / 2);
    sA[j] = pk2(g[(size_t)r * gstride + 2 * cp],
                g[(size_t)r * gstride + 2 * cp + 1]);
  }
}

// Butterfly reductions within a 16-lane half (xor mask < 16 stays in half).
DEV float half_max(float x) {
#pragma unroll
  for (int d = 1; d < 16; d <<= 1) x = fmaxf(x, __shfl_xor(x, d, 32));
  return x;
}
DEV float half_sum(float x) {
#pragma unroll
  for (int d = 1; d < 16; d <<= 1) x += __shfl_xor(x, d, 32);
  return x;
}

// ---------------------------------------------------------------------------
// Tensor Data Mover: async 2-D f32 tile global -> LDS.  D# built per ISA ch.8.
// ---------------------------------------------------------------------------
DEV void tdm_load_2d(unsigned int lds_off, const float* gptr, unsigned int tile_w,
                     unsigned int tile_h, unsigned int stride_elems) {
  unsigned long long ga = (unsigned long long)gptr;
  u32x4 g0;
  g0[0] = 1u;                                                     // count=1
  g0[1] = lds_off;                                                // lds_addr
  g0[2] = (unsigned int)ga;                                       // gaddr[31:0]
  g0[3] = (unsigned int)((ga >> 32) & 0x01FFFFFFu) | (2u << 30);  // [56:32]|type=2
  i32x8 g1;
  g1[0] = (int)(2u << 16);                                        // data_size=4B
  g1[1] = (int)((tile_w & 0xFFFFu) << 16);                        // tensor_dim0 lo
  g1[2] = (int)(((tile_w >> 16) & 0xFFFFu) | ((tile_h & 0xFFFFu) << 16));
  g1[3] = (int)(((tile_h >> 16) & 0xFFFFu) | (tile_w << 16));     // tile_dim0
  g1[4] = (int)tile_h;                                            // tile_dim1
  g1[5] = (int)stride_elems;                                      // dim0_stride lo
  g1[6] = 0;
  g1[7] = 0;
  i32x4 z = {};
#if defined(__clang_major__) && (__clang_major__ >= 23)
  i32x8 z8 = {};
  __builtin_amdgcn_tensor_load_to_lds(g0, g1, z, z, z8, 0);
#else
  __builtin_amdgcn_tensor_load_to_lds(g0, g1, z, z, 0);
#endif
}

// ============================================================================
// 1) GNN layer. grid (N/16, B), block 128 (wave w owns output cols 64w..64w+63)
// ============================================================================
__global__ __launch_bounds__(128)
void gnn_layer_kernel(const float* __restrict__ h_in, const float* __restrict__ adj,
                      const float* __restrict__ W, const float* __restrict__ bias,
                      const float* __restrict__ gamma, const float* __restrict__ beta,
                      float* __restrict__ h_out) {
  __shared__ unsigned int sAdj[16 * 256];   // 16x512 f16 row-major (dwords)
  __shared__ unsigned int sB[16 * 256];     // 32x256 f16 PK2
  __shared__ f16          sMsg[16 * 256];
  __shared__ float        sOut[16 * 257];
  __shared__ float        sRed[16][8][2];
  __shared__ float        sStat[16][2];

  const int tid  = threadIdx.x;
  const int wave = tid >> 5, lane = tid & 31;
  const int n0 = blockIdx.x * 16;
  const int b  = blockIdx.y;
  const float* hB = h_in + (size_t)b * N_ * D_;
  const float* aB = adj  + (size_t)b * N_ * N_;

  stage_a<512>(sAdj, aB + (size_t)n0 * N_, N_, 16, tid, 128);

  v8f acc[4] = {};
  for (int kt = 0; kt < N_; kt += 32) {
    stage_b_pk2<256>(sB, hB + (size_t)kt * D_, D_, 32, tid, 128);
    __syncthreads();
    v16h aF = load_a((const f16*)sAdj + kt, 512, lane);
#pragma unroll
    for (int t = 0; t < 4; ++t)
      acc[t] = wmma16(aF, load_b_pk2(sB + wave * 64 + t * 16, 256, lane), acc[t]);
    __syncthreads();
  }
  {
    int r8 = 8 * (lane >> 4), c0 = lane & 15;
#pragma unroll
    for (int t = 0; t < 4; ++t)
#pragma unroll
      for (int v = 0; v < 8; ++v)
        sMsg[(v + r8) * 256 + wave * 64 + t * 16 + c0] = (f16)acc[t][v];
  }
  __syncthreads();

  v8f acc2[4] = {};
  for (int kt = 0; kt < D_; kt += 32) {
    stage_b_pk2<256>(sB, W + (size_t)kt * D_, D_, 32, tid, 128);
    __syncthreads();
    v16h aF = load_a(sMsg + kt, 256, lane);
#pragma unroll
    for (int t = 0; t < 4; ++t)
      acc2[t] = wmma16(aF, load_b_pk2(sB + wave * 64 + t * 16, 256, lane), acc2[t]);
    __syncthreads();
  }
  {
    int r8 = 8 * (lane >> 4), c0 = lane & 15;
#pragma unroll
    for (int t = 0; t < 4; ++t)
#pragma unroll
      for (int v = 0; v < 8; ++v) {
        int r = v + r8, c = wave * 64 + t * 16 + c0;
        float x = acc2[t][v] + bias[c];
        x = x > 0.f ? x : 0.f;
        x += hB[(size_t)(n0 + r) * D_ + c];
        sOut[r * 257 + c] = x;
      }
  }
  __syncthreads();
  {
    int r = tid >> 3, seg = tid & 7;
    float s = 0.f, sq = 0.f;
    for (int c = seg * 32; c < seg * 32 + 32; ++c) {
      float x = sOut[r * 257 + c];
      s += x; sq += x * x;
    }
    sRed[r][seg][0] = s; sRed[r][seg][1] = sq;
  }
  __syncthreads();
  if (tid < 16) {
    float s = 0.f, sq = 0.f;
    for (int g = 0; g < 8; ++g) { s += sRed[tid][g][0]; sq += sRed[tid][g][1]; }
    float m   = s * (1.f / 256.f);
    float var = sq * (1.f / 256.f) - m * m;
    sStat[tid][0] = m;
    sStat[tid][1] = rsqrtf(var + 1e-3f);
  }
  __syncthreads();
  for (int i = tid; i < 16 * 256; i += 128) {
    int r = i >> 8, c = i & 255;
    float x = (sOut[r * 257 + c] - sStat[r][0]) * sStat[r][1] * gamma[c] + beta[c];
    h_out[(size_t)b * N_ * D_ + (size_t)(n0 + r) * D_ + c] = x;
  }
}

// ============================================================================
// 2) q/k/v projection (v stored transposed [b][c][n]). grid (N/16, B, 3).
// ============================================================================
__global__ __launch_bounds__(128)
void qkv_kernel(const float* __restrict__ h,
                const float* __restrict__ Wq, const float* __restrict__ Wk,
                const float* __restrict__ Wv, const float* __restrict__ bq,
                const float* __restrict__ bk, const float* __restrict__ bv,
                f16* __restrict__ q, f16* __restrict__ k, f16* __restrict__ vT) {
  const float* W; const float* bb;
  if (blockIdx.z == 0)      { W = Wq; bb = bq; }
  else if (blockIdx.z == 1) { W = Wk; bb = bk; }
  else                      { W = Wv; bb = bv; }

  __shared__ unsigned int sA[16 * 128];
  __shared__ unsigned int sB[16 * 256];
  const int tid  = threadIdx.x;
  const int wave = tid >> 5, lane = tid & 31;
  const int n0 = blockIdx.x * 16;
  const int b  = blockIdx.y;
  const float* hB = h + ((size_t)b * N_ + n0) * D_;

  stage_a<256>(sA, hB, D_, 16, tid, 128);

  v8f acc[4] = {};
  for (int kt = 0; kt < D_; kt += 32) {
    stage_b_pk2<256>(sB, W + (size_t)kt * 256, 256, 32, tid, 128);
    __syncthreads();
    v16h aF = load_a((const f16*)sA + kt, 256, lane);
#pragma unroll
    for (int t = 0; t < 4; ++t)
      acc[t] = wmma16(aF, load_b_pk2(sB + wave * 64 + t * 16, 256, lane), acc[t]);
    __syncthreads();
  }
  int r8 = 8 * (lane >> 4), c0 = lane & 15;
#pragma unroll
  for (int t = 0; t < 4; ++t)
#pragma unroll
    for (int v = 0; v < 8; ++v) {
      int r = v + r8, c = wave * 64 + t * 16 + c0;
      f16 val = (f16)(acc[t][v] + bb[c]);
      if (blockIdx.z == 2)
        vT[((size_t)b * 256 + c) * N_ + n0 + r] = val;
      else if (blockIdx.z == 0)
        q[((size_t)b * N_ + n0 + r) * 256 + c] = val;
      else
        k[((size_t)b * N_ + n0 + r) * 256 + c] = val;
    }
}

// ============================================================================
// 3) Flash attention. grid (N/64, H, B), block 128; wave owns 16 query rows.
// ============================================================================
__global__ __launch_bounds__(128)
void attn_kernel(const f16* __restrict__ q, const f16* __restrict__ k,
                 const f16* __restrict__ vT, f16* __restrict__ ctx) {
  __shared__ f16 sP[4][16 * 32];
  const int tid  = threadIdx.x;
  const int wave = tid >> 5, lane = tid & 31;
  const int head = blockIdx.y, b = blockIdx.z;
  const int n0 = blockIdx.x * 64 + wave * 16;
  const int c0 = lane & 15, hi = lane >> 4;

  const f16* qp = q + ((size_t)(b * N_ + n0) * 256 + head * 64);
  v16h aQ0 = load_a(qp, 256, lane);
  v16h aQ1 = load_a(qp + 32, 256, lane);

  v8f o[4] = {};
  float mrun[8], lrun[8];
#pragma unroll
  for (int vv = 0; vv < 8; ++vv) { mrun[vv] = -1e30f; lrun[vv] = 0.f; }

  for (int s0 = 0; s0 < N_; s0 += 32) {
    const f16* kp0 = k + ((size_t)(b * N_ + s0) * 256 + head * 64);
    const f16* kp1 = kp0 + 16 * 256;
    v8f slo = {}, shi = {};
    slo = wmma16(aQ0, load_bT(kp0,      256, lane), slo);
    slo = wmma16(aQ1, load_bT(kp0 + 32, 256, lane), slo);
    shi = wmma16(aQ0, load_bT(kp1,      256, lane), shi);
    shi = wmma16(aQ1, load_bT(kp1 + 32, 256, lane), shi);
#pragma unroll
    for (int vv = 0; vv < 8; ++vv) {
      float xl = slo[vv] * 0.125f, xh = shi[vv] * 0.125f;
      float mv   = half_max(fmaxf(xl, xh));
      float mnew = fmaxf(mrun[vv], mv);
      float alpha = __expf(mrun[vv] - mnew);
      float pl = __expf(xl - mnew), ph = __expf(xh - mnew);
      lrun[vv] = lrun[vv] * alpha + half_sum(pl + ph);
      mrun[vv] = mnew;
#pragma unroll
      for (int t = 0; t < 4; ++t) o[t][vv] *= alpha;
      sP[wave][(vv + 8 * hi) * 32 + c0]      = (f16)pl;
      sP[wave][(vv + 8 * hi) * 32 + c0 + 16] = (f16)ph;
    }
    __syncthreads();
    v16h aP = load_a(&sP[wave][0], 32, lane);
    const f16* vp = vT + ((size_t)b * 256 + head * 64) * N_ + s0;
#pragma unroll
    for (int t = 0; t < 4; ++t)
      o[t] = wmma16(aP, load_bT(vp + (size_t)t * 16 * N_, N_, lane), o[t]);
    __syncthreads();
  }
#pragma unroll
  for (int vv = 0; vv < 8; ++vv) {
    float inv = 1.f / lrun[vv];
#pragma unroll
    for (int t = 0; t < 4; ++t)
      ctx[((size_t)(b * N_ + n0 + vv + 8 * hi)) * 256 + head * 64 + t * 16 + c0] =
          (f16)(o[t][vv] * inv);
  }
}

// ============================================================================
// 4) Output projection. grid (N/16, B), block 128.
// ============================================================================
__global__ __launch_bounds__(128)
void proj_kernel(const f16* __restrict__ ctx, const float* __restrict__ Wo,
                 const float* __restrict__ bo, float* __restrict__ out) {
  __shared__ unsigned int sB[16 * 256];
  const int tid  = threadIdx.x;
  const int wave = tid >> 5, lane = tid & 31;
  const int n0 = blockIdx.x * 16;
  const int b  = blockIdx.y;
  const f16* ap = ctx + ((size_t)b * N_ + n0) * 256;

  v8f acc[4] = {};
  for (int kt = 0; kt < 256; kt += 32) {
    stage_b_pk2<256>(sB, Wo + (size_t)kt * 256, 256, 32, tid, 128);
    __syncthreads();
    v16h aF = load_a(ap + kt, 256, lane);
#pragma unroll
    for (int t = 0; t < 4; ++t)
      acc[t] = wmma16(aF, load_b_pk2(sB + wave * 64 + t * 16, 256, lane), acc[t]);
    __syncthreads();
  }
  int r8 = 8 * (lane >> 4), c0 = lane & 15;
#pragma unroll
  for (int t = 0; t < 4; ++t)
#pragma unroll
    for (int v = 0; v < 8; ++v) {
      int r = v + r8, c = wave * 64 + t * 16 + c0;
      out[((size_t)b * N_ + n0 + r) * 256 + c] = acc[t][v] + bo[c];
    }
}

// ============================================================================
// 5) context mean + query projection (f16 out). grid B, block 256.
// ============================================================================
__global__ __launch_bounds__(256)
void query_kernel(const float* __restrict__ h, const float* __restrict__ Wq,
                  const float* __restrict__ bq, f16* __restrict__ qf16) {
  __shared__ float sCtx[256];
  const int b = blockIdx.x, tid = threadIdx.x;
  float s = 0.f;
  for (int n = 0; n < N_; ++n) s += h[((size_t)b * N_ + n) * D_ + tid];
  sCtx[tid] = s * (1.f / N_);
  __syncthreads();
  float acc = bq[tid];
  for (int d = 0; d < D_; ++d) acc += sCtx[d] * Wq[(size_t)d * KD_ + tid];
  qf16[b * KD_ + tid] = (f16)acc;
}

// ============================================================================
// 6) Logits + per-split stats. grid 512 splits x 512 slots, block 256.
//    Keys streamed ONCE via double-buffered TDM slabs.
// ============================================================================
__global__ __launch_bounds__(256)
void logits_kernel(const f16* __restrict__ qf, const float* __restrict__ keys,
                   f16* __restrict__ logits, float* __restrict__ stats) {
  __shared__ float sK[2][32 * 256];   // 64 KB double buffer
  const int tid  = threadIdx.x;
  const int wave = tid >> 5, lane = tid & 31;
  const int sp = blockIdx.x;
  const int slot0 = sp * 512;
  const int c0 = lane & 15, hi = lane >> 4;

  float mrun[2][8], lrun[2][8];
#pragma unroll
  for (int rt = 0; rt < 2; ++rt)
#pragma unroll
    for (int vv = 0; vv < 8; ++vv) { mrun[rt][vv] = -1e30f; lrun[rt][vv] = 0.f; }

  if (tid < 32)
    tdm_load_2d((unsigned int)(size_t)&sK[0][0],
                keys + (size_t)slot0 * KD_, KD_, 32, KD_);

  for (int st = 0; st < 512; st += 32) {
    const int cur = (st >> 5) & 1;
    if (tid < 32) {
      if (st + 32 < 512) {
        tdm_load_2d((unsigned int)(size_t)&sK[cur ^ 1][0],
                    keys + (size_t)(slot0 + st + 32) * KD_, KD_, 32, KD_);
        __builtin_amdgcn_s_wait_tensorcnt(1);
      } else {
        __builtin_amdgcn_s_wait_tensorcnt(0);
      }
    }
    __syncthreads();
    const float* sKc = &sK[cur][0];
#pragma unroll
    for (int rt = 0; rt < 2; ++rt) {
      const int row0 = wave * 32 + rt * 16;
      const f16* qp = qf + (size_t)row0 * KD_;
      v16h aQ[8];
#pragma unroll
      for (int k8 = 0; k8 < 8; ++k8) aQ[k8] = load_a(qp + k8 * 32, KD_, lane);
#pragma unroll
      for (int ct = 0; ct < 2; ++ct) {
        v8f acc = {};
#pragma unroll
        for (int k8 = 0; k8 < 8; ++k8)
          acc = wmma16(aQ[k8],
                       load_bT_f32(sKc + (ct * 16) * 256 + k8 * 32, 256, lane), acc);
#pragma unroll
        for (int vv = 0; vv < 8; ++vv) {
          float val = acc[vv];
          float mv   = half_max(val);
          float mnew = fmaxf(mrun[rt][vv], mv);
          float alpha = __expf(mrun[rt][vv] - mnew);
          lrun[rt][vv] = lrun[rt][vv] * alpha + half_sum(__expf(val - mnew));
          mrun[rt][vv] = mnew;
          logits[(size_t)(row0 + vv + 8 * hi) * S_ + slot0 + st + ct * 16 + c0] =
              (f16)val;
        }
      }
    }
    __syncthreads();
  }
  if (c0 == 0) {
#pragma unroll
    for (int rt = 0; rt < 2; ++rt)
#pragma unroll
      for (int vv = 0; vv < 8; ++vv) {
        int row = wave * 32 + rt * 16 + vv + 8 * hi;
        stats[((size_t)sp * 256 + row) * 2 + 0] = mrun[rt][vv];
        stats[((size_t)sp * 256 + row) * 2 + 1] = lrun[rt][vv];
      }
  }
}

// ============================================================================
// 7) Combine split stats. grid 1, block 256.
// ============================================================================
__global__ __launch_bounds__(256)
void stats_combine_kernel(const float* __restrict__ stats, float* __restrict__ ML) {
  const int r = threadIdx.x;
  float M = -1e30f;
  for (int sp = 0; sp < 512; ++sp)
    M = fmaxf(M, stats[((size_t)sp * 256 + r) * 2]);
  float L = 0.f;
  for (int sp = 0; sp < 512; ++sp)
    L += stats[((size_t)sp * 256 + r) * 2 + 1] *
         __expf(stats[((size_t)sp * 256 + r) * 2] - M);
  ML[r * 2]     = M;
  ML[r * 2 + 1] = 1.f / L;
}

// ============================================================================
// 8) part = softmax(logits) @ values. grid (4, 64), block 1024 (32 waves).
//    Values read ONCE via double-buffered TDM slabs.
// ============================================================================
__global__ __launch_bounds__(1024)
void retrieve_kernel(const f16* __restrict__ logits, const float* __restrict__ vals,
                     const float* __restrict__ ML, float* __restrict__ part) {
  __shared__ float sV[2][32 * 128];   // 32 KB double buffer
  const int tid  = threadIdx.x;
  const int wave = tid >> 5, lane = tid & 31;
  const int cg = blockIdx.x;
  const int ks = blockIdx.y;
  const int rt = wave >> 1;
  const int cs = (wave & 1) * 64;
  const int row = rt * 16 + (lane & 15);
  const int kh  = (lane >> 4) * 8;    // pair units
  const float Mr   = ML[row * 2];
  const float invL = ML[row * 2 + 1];
  const unsigned int* lp =
      (const unsigned int*)(logits + (size_t)row * S_ + (size_t)ks * 4096);

  if (tid < 32)
    tdm_load_2d((unsigned int)(size_t)&sV[0][0],
                vals + (size_t)(ks * 4096) * MD_ + cg * 128, 128, 32, MD_);

  v8f acc[4] = {};
  for (int kt = 0; kt < 4096; kt += 32) {
    const int cur = (kt >> 5) & 1;
    if (tid < 32) {
      if (kt + 32 < 4096) {
        tdm_load_2d((unsigned int)(size_t)&sV[cur ^ 1][0],
                    vals + (size_t)(ks * 4096 + kt + 32) * MD_ + cg * 128,
                    128, 32, MD_);
        __builtin_amdgcn_s_wait_tensorcnt(1);
      } else {
        __builtin_amdgcn_s_wait_tensorcnt(0);
      }
    }
    __syncthreads();
    v16h aW;
    {
      v8u u;
#pragma unroll
      for (int v = 0; v < 8; ++v) {
        int kp = kh + v;
        h2 pr = __builtin_bit_cast(h2, lp[(kt >> 1) + kp]);
        u[v] = pk2(__expf((float)pr.x - Mr) * invL,
                   __expf((float)pr.y - Mr) * invL);
      }
      aW = __builtin_bit_cast(v16h, u);
    }
    const float* sVc = &sV[cur][0];
#pragma unroll
    for (int t = 0; t < 4; ++t)
      acc[t] = wmma16(aW, load_b_f32(sVc + cs + t * 16, 128, lane), acc[t]);
    __syncthreads();
  }
  const int c0 = lane & 15, hi = lane >> 4;
  float* pp = part + (((size_t)cg * 64 + ks) * 256) * 128;
#pragma unroll
  for (int t = 0; t < 4; ++t)
#pragma unroll
    for (int v = 0; v < 8; ++v)
      pp[(size_t)(rt * 16 + v + 8 * hi) * 128 + cs + t * 16 + c0] = acc[t][v];
}

// ============================================================================
// 9) split-K fixup. grid 512, block 256 -> d_out f32 [256, 512].
// ============================================================================
__global__ __launch_bounds__(256)
void combine_kernel(const float* __restrict__ part, float* __restrict__ out) {
  const int idx = blockIdx.x * 256 + threadIdx.x;
  const int r = idx >> 9, c = idx & 511;
  const int cg = c >> 7, cc = c & 127;
  float s = 0.f;
  for (int ks = 0; ks < 64; ++ks)
    s += part[(((size_t)cg * 64 + ks) * 256 + r) * 128 + cc];
  out[idx] = s;
}

// ============================================================================
extern "C" void kernel_launch(void* const* d_in, const int* in_sizes, int n_in,
                              void* d_out, int out_size, void* d_ws, size_t ws_size,
                              hipStream_t stream) {
  (void)in_sizes; (void)n_in; (void)out_size; (void)ws_size;
  const float* nf   = (const float*)d_in[0];
  const float* adj  = (const float*)d_in[1];
  const float* keys = (const float*)d_in[2];
  const float* vals = (const float*)d_in[3];
  const float* gW   = (const float*)d_in[4];
  const float* gb   = (const float*)d_in[5];
  const float* lg   = (const float*)d_in[6];
  const float* lb   = (const float*)d_in[7];
  const float* Wq   = (const float*)d_in[8];
  const float* bq   = (const float*)d_in[9];
  const float* Wk   = (const float*)d_in[10];
  const float* bk   = (const float*)d_in[11];
  const float* Wv   = (const float*)d_in[12];
  const float* bv   = (const float*)d_in[13];
  const float* Wo   = (const float*)d_in[14];
  const float* bo   = (const float*)d_in[15];
  const float* Wqr  = (const float*)d_in[16];
  const float* bqr  = (const float*)d_in[17];

  const size_t MB = (size_t)1 << 20;
  char* ws = (char*)d_ws;
  float* hA    = (float*)(ws);                  // 128 MB
  float* hBf   = (float*)(ws + 128 * MB);       // 128 MB
  f16*   qb    = (f16*)(ws + 256 * MB);         // 64 MB
  f16*   kb    = (f16*)(ws + 320 * MB);         // 64 MB
  f16*   vTb   = (f16*)(ws + 384 * MB);         // 64 MB (transposed v)
  f16*   ctxb  = (f16*)(ws + 448 * MB);         // 64 MB
  f16*   logit = (f16*)(ws + 256 * MB);         // 128 MB (aliases q/k, dead)
  float* part  = (float*)(ws + 448 * MB);       // 32 MB  (aliases ctx, dead)
  f16*   qf16  = (f16*)(ws + 512 * MB);         // 128 KB
  float* stats = (float*)(ws + 513 * MB);       // 1 MB
  float* ML    = (float*)(ws + 514 * MB);       // 2 KB

  dim3 gGemm(32, 256);
  gnn_layer_kernel<<<gGemm, 128, 0, stream>>>(nf,  adj, gW + 0 * D_ * D_, gb + 0 * D_,
                                              lg + 0 * D_, lb + 0 * D_, hBf);
  gnn_layer_kernel<<<gGemm, 128, 0, stream>>>(hBf, adj, gW + 1 * D_ * D_, gb + 1 * D_,
                                              lg + 1 * D_, lb + 1 * D_, hA);
  gnn_layer_kernel<<<gGemm, 128, 0, stream>>>(hA,  adj, gW + 2 * D_ * D_, gb + 2 * D_,
                                              lg + 2 * D_, lb + 2 * D_, hBf);

  qkv_kernel<<<dim3(32, 256, 3), 128, 0, stream>>>(hBf, Wq, Wk, Wv, bq, bk, bv,
                                                   qb, kb, vTb);
  attn_kernel<<<dim3(8, 4, 256), 128, 0, stream>>>(qb, kb, vTb, ctxb);
  proj_kernel<<<gGemm, 128, 0, stream>>>(ctxb, Wo, bo, hA);
  query_kernel<<<256, 256, 0, stream>>>(hA, Wqr, bqr, qf16);

  logits_kernel<<<512, 256, 0, stream>>>(qf16, keys, logit, stats);
  stats_combine_kernel<<<1, 256, 0, stream>>>(stats, ML);
  retrieve_kernel<<<dim3(4, 64), 1024, 0, stream>>>(logit, vals, ML, part);
  combine_kernel<<<512, 256, 0, stream>>>(part, (float*)d_out);
}
